// HeteroEGIN_32882269618406
// MI455X (gfx1250) — matrix operand order repaired and checked by reading, not compiled, not run on hardware
//
#include <hip/hip_runtime.h>
#include <hip/hip_bf16.h>

// ---------------------------------------------------------------------------
// CDNA5 (gfx1250) HeteroEGIN: bf16 WMMA for all GEMMs (f32 accumulate),
// TDM (tensor_load_to_lds) staging of packed weights into LDS,
// atomic f32 scatter for segment_sum.
// ---------------------------------------------------------------------------

typedef __attribute__((ext_vector_type(16))) __bf16 v16bf;
typedef __attribute__((ext_vector_type(8)))  float  v8f;
typedef unsigned int u32x4 __attribute__((ext_vector_type(4)));
typedef int          i32x8 __attribute__((ext_vector_type(8)));
typedef int          i32x4 __attribute__((ext_vector_type(4)));

struct __align__(16) U4 { unsigned int x, y, z, w; };

union Frag16 {
    v16bf          v;
    unsigned short u[16];
    U4             q[2];
};

__device__ __forceinline__ unsigned short f2bf(float f) {
    unsigned int u = __float_as_uint(f);
    u += 0x7FFFu + ((u >> 16) & 1u);          // round-to-nearest-even
    return (unsigned short)(u >> 16);
}

__device__ __forceinline__ v8f wmma_bf16(const Frag16& a, const Frag16& b, v8f c) {
    return __builtin_amdgcn_wmma_f32_16x16x32_bf16(
        /*neg_a=*/false, a.v, /*neg_b=*/false, b.v,
        /*c_mod=*/(short)0, c, /*reuse_a=*/false, /*reuse_b=*/false);
}

// ---------------------------------------------------------------------------
// TDM: DMA `nelem` bf16 (2-byte) elements from global into LDS at byte offset
// `lds_byte_off` within this workgroup's LDS allocation. 1-D tile descriptor
// per CDNA5 ISA section 8.3/8.4. Issue once per wave; wait with TENSORcnt.
// (6-arg builtin form: g0 v4u, g1 v8i, g2 v4i, g3 v4i, extra v8i, cpol)
// ---------------------------------------------------------------------------
__device__ __forceinline__ void tdm_load_bf16_1d(unsigned lds_byte_off,
                                                 const unsigned short* src,
                                                 unsigned nelem) {
    unsigned long long ga = (unsigned long long)(const void*)src;
    u32x4 g0;
    g0[0] = 1u;                                          // count=1 (valid user D#)
    g0[1] = __builtin_amdgcn_groupstaticsize() + lds_byte_off;  // lds_addr
    g0[2] = (unsigned)ga;                                // global_addr[31:0]
    g0[3] = (unsigned)(ga >> 32) | (2u << 30);           // global_addr[56:32] | type=2
    i32x8 g1;
    g1[0] = 1 << 16;                                     // data_size = 1 -> 2 bytes
    g1[1] = (int)((nelem & 0xFFFFu) << 16);              // tensor_dim0[15:0]
    g1[2] = (int)((nelem >> 16) | (1u << 16));           // tensor_dim0[31:16], tensor_dim1=1
    g1[3] = (int)((nelem & 0xFFFFu) << 16);              // tile_dim0 = nelem
    g1[4] = 0;                                           // tile_dim1/2 unused
    g1[5] = (int)nelem;                                  // tensor_dim0_stride[31:0]
    g1[6] = 0;
    g1[7] = 0;
    i32x4 gz4 = {0, 0, 0, 0};
    i32x8 gz8 = {0, 0, 0, 0, 0, 0, 0, 0};
    __builtin_amdgcn_tensor_load_to_lds(g0, g1, gz4, gz4, gz8, 0);
}

#define RBF_START    0.006737946999085467f               /* exp(-5) */
#define RBF_STEP     ((1.0f - RBF_START) / 127.0f)
#define RBF_BETA     (1.0f / ((2.0f/128.0f*(1.0f-RBF_START)) * (2.0f/128.0f*(1.0f-RBF_START))))
#define PI_OVER_DCUT 0.6283185307179586f                 /* pi / 5 */
#define D_CUT_F      5.0f

// ---------------------------------------------------------------------------
// Pack a row-major f32 weight [slices][K][Ncols] into bf16 WMMA B-fragment
// order: out[((kt*NT+nt)*32 + lane)*16 + e] = W[32kt + 8*(lane>>4) + (e&7) + 16*(e>>3)][16nt + (lane&15)]
// ---------------------------------------------------------------------------
__global__ void pack_wmma_b(const float* __restrict__ W, unsigned short* __restrict__ out,
                            int K, int Ncols, int slices) {
    const int  KT = K >> 5, NT = Ncols >> 4;
    const long per   = (long)KT * NT * 512;
    const long total = per * slices;
    long idx    = (long)blockIdx.x * blockDim.x + threadIdx.x;
    long stride = (long)gridDim.x * blockDim.x;
    for (; idx < total; idx += stride) {
        int  s    = (int)(idx / per);
        long r    = idx % per;
        int  tile = (int)(r >> 9);
        int  w    = (int)(r & 511);
        int  lane = w >> 4;
        int  e    = w & 15;
        int  kt   = tile / NT, nt = tile % NT;
        int  row  = kt * 32 + 8 * (lane >> 4) + (e & 7) + 16 * (e >> 3);
        int  col  = nt * 16 + (lane & 15);
        out[idx]  = f2bf(W[((long)s * K + row) * Ncols + col]);
    }
}

__global__ void zero_kernel(float* __restrict__ p, long n) {
    long i      = (long)blockIdx.x * blockDim.x + threadIdx.x;
    long stride = (long)gridDim.x * blockDim.x;
    for (; i < n; i += stride) p[i] = 0.0f;
}

// ---------------------------------------------------------------------------
// Edge kernel: per wave, 16 edges.
//   ss[16x256] = rbf(ew)[16x128] @ Wd[128x256] + bd      (WMMA, bf16 in / f32 acc)
//   msg = relu((x_src[ei0] + ea@W0@We + be) * scale + shift)
//   atomicAdd into agg[ei1]
// Wd pack (64KB) staged in LDS once per block via TDM.
// Dynamic LDS: 64KB weights + 3KB edge metadata = 68608 B.
// ---------------------------------------------------------------------------
__global__ __launch_bounds__(256) void edge_conv_kernel(
    const float* __restrict__ ew, const int* __restrict__ ei,
    const float* __restrict__ ea, const float* __restrict__ x_src,
    const unsigned short* __restrict__ WdP, const float* __restrict__ bd,
    const float* __restrict__ W0t, const float* __restrict__ We,
    const float* __restrict__ be, float* __restrict__ agg, int E)
{
    extern __shared__ char smem[];
    unsigned short* sWd = (unsigned short*)smem;          // 32768 bf16 = 64KB
    int*   sJ  = (int*)(smem + 65536);                    // [8][16]
    int*   sD  = sJ + 128;                                // [8][16]
    float* sA4 = (float*)(sD + 128);                      // [8][16][4]

    const int tid  = threadIdx.x;
    const int wave = tid >> 5;
    const int lane = tid & 31;
    const int half = lane >> 4;
    const int m    = lane & 15;

    if (wave == 0) {                                      // one TDM per block
        tdm_load_bf16_1d(0, WdP, 32768);
        __builtin_amdgcn_s_wait_tensorcnt(0);
    }

    const int  tileBase = blockIdx.x * 128 + wave * 16;
    const int  e        = tileBase + m;
    const bool valid    = (e < E);
    const int  ec       = valid ? e : 0;

    const float d   = ew[ec];
    const float a0i = ea[(long)ec*4+0], a1i = ea[(long)ec*4+1];
    const float a2i = ea[(long)ec*4+2], a3i = ea[(long)ec*4+3];

    if (lane < 16) {
        sJ[wave*16 + m] = valid ? ei[ec]     : 0;
        sD[wave*16 + m] = valid ? ei[E + ec] : 0;
        #pragma unroll
        for (int c = 0; c < 4; ++c)
            sA4[(wave*16 + m)*4 + c] =
                a0i*W0t[0*4+c] + a1i*W0t[1*4+c] + a2i*W0t[2*4+c] + a3i*W0t[3*4+c];
    }
    __syncthreads();

    // RBF features directly into A-fragment layout (4 K-tiles of 32)
    const float dc     = fminf(d, D_CUT_F);
    const float cutoff = valid ? 0.5f * (__cosf(dc * PI_OVER_DCUT) + 1.0f) : 0.0f;
    const float emd    = __expf(-d);

    Frag16 afr[4];
    #pragma unroll
    for (int kt = 0; kt < 4; ++kt) {
        #pragma unroll
        for (int e2 = 0; e2 < 16; ++e2) {
            int   K    = kt*32 + (e2 & 7) + 8*half + 16*(e2 >> 3);
            float mean = RBF_START + (float)K * RBF_STEP;
            float t    = emd - mean;
            afr[kt].u[e2] = f2bf(cutoff * __expf(-RBF_BETA * t * t));
        }
    }

    const bool fullTile = (tileBase + 16 <= E);

    for (int nt = 0; nt < 8; ++nt) {
        const int   hcol = nt*16 + m;
        const float bs   = bd[hcol];
        const float bh   = bd[128 + hcol];
        v8f cs = {bs,bs,bs,bs,bs,bs,bs,bs};   // scale tile (cols nt)
        v8f ch = {bh,bh,bh,bh,bh,bh,bh,bh};   // shift tile (cols nt+8)
        #pragma unroll
        for (int kt = 0; kt < 4; ++kt) {
            Frag16 b0, b1;
            const U4* p0 = (const U4*)&sWd[(((kt*16 + nt    ) * 32) + lane) * 16];
            const U4* p1 = (const U4*)&sWd[(((kt*16 + nt + 8) * 32) + lane) * 16];
            b0.q[0] = p0[0]; b0.q[1] = p0[1];
            b1.q[0] = p1[0]; b1.q[1] = p1[1];
            cs = wmma_bf16(afr[kt], b0, cs);
            ch = wmma_bf16(afr[kt], b1, ch);
        }

        const float we0 = We[0*128+hcol], we1 = We[1*128+hcol];
        const float we2 = We[2*128+hcol], we3 = We[3*128+hcol];
        const float beh = be[hcol];

        if (fullTile) {
            #pragma unroll
            for (int v = 0; v < 8; ++v) {
                int mr = v + 8*half;
                int j  = sJ[wave*16 + mr];
                int dr = sD[wave*16 + mr];
                const float* a4 = &sA4[(wave*16 + mr)*4];
                float xj  = x_src[(long)j*128 + hcol]
                          + a4[0]*we0 + a4[1]*we1 + a4[2]*we2 + a4[3]*we3 + beh;
                float msg = fmaxf(xj * cs[v] + ch[v], 0.0f);
                atomicAdd(&agg[(long)dr*128 + hcol], msg);
            }
        } else {
            #pragma unroll
            for (int v = 0; v < 8; ++v) {
                int mr = v + 8*half;
                int eg = tileBase + mr;
                int j  = sJ[wave*16 + mr];
                int dr = sD[wave*16 + mr];
                const float* a4 = &sA4[(wave*16 + mr)*4];
                float xj  = x_src[(long)j*128 + hcol]
                          + a4[0]*we0 + a4[1]*we1 + a4[2]*we2 + a4[3]*we3 + beh;
                float msg = fmaxf(xj * cs[v] + ch[v], 0.0f);
                if (eg < E) atomicAdd(&agg[(long)dr*128 + hcol], msg);
            }
        }
    }
}

// ---------------------------------------------------------------------------
// Node kernel: per wave, 16 rows.
//   o  = agg + x_dst          (bf16 into LDS)
//   h  = relu(o @ W1 + b1)    (WMMA)
//   y  = h @ W2 + b2          (WMMA)
//   out = [accumulate? out+y : y]; optional relu after accumulation
// W1/W2 packs (2x32KB) staged in LDS via TDM; per-wave 4KB o/h tile.
// Dynamic LDS: 64KB weights + 32KB tiles = 98304 B.
// ---------------------------------------------------------------------------
__global__ __launch_bounds__(256) void node_mlp_kernel(
    const float* __restrict__ agg, const float* __restrict__ xdst,
    const unsigned short* __restrict__ W1P, const float* __restrict__ b1,
    const unsigned short* __restrict__ W2P, const float* __restrict__ b2,
    float* __restrict__ out, int N, int accumulate, int doRelu)
{
    extern __shared__ char smem[];
    unsigned short* sW1 = (unsigned short*)smem;          // 16384 bf16
    unsigned short* sW2 = sW1 + 16384;                    // 16384 bf16
    unsigned short* sOb = sW2 + 16384;                    // [8][16*128] bf16

    const int tid  = threadIdx.x;
    const int wave = tid >> 5;
    const int lane = tid & 31;
    const int half = lane >> 4;
    const int m    = lane & 15;
    const int rowBase = blockIdx.x * 128 + wave * 16;
    unsigned short* sO = sOb + wave * 2048;

    if (wave == 0) {
        tdm_load_bf16_1d(0,     W1P, 16384);
        tdm_load_bf16_1d(32768, W2P, 16384);
        __builtin_amdgcn_s_wait_tensorcnt(0);
    }

    { // stage o = agg + xdst (row m, 64-col half per lane)
        int r  = rowBase + m;
        int rc = (r < N) ? r : (N - 1);
        const float4* ap = (const float4*)(agg  + (long)rc*128 + half*64);
        const float4* xp = (const float4*)(xdst + (long)rc*128 + half*64);
        unsigned short* op = &sO[m*128 + half*64];
        #pragma unroll
        for (int i = 0; i < 16; ++i) {
            float4 a = ap[i]; float4 x = xp[i];
            op[i*4+0] = f2bf(a.x + x.x);
            op[i*4+1] = f2bf(a.y + x.y);
            op[i*4+2] = f2bf(a.z + x.z);
            op[i*4+3] = f2bf(a.w + x.w);
        }
    }
    __syncthreads();

    // GEMM1: h = relu(o @ W1 + b1)
    v8f h[8];
    for (int nt = 0; nt < 8; ++nt) {
        float bb = b1[nt*16 + m];
        v8f c = {bb,bb,bb,bb,bb,bb,bb,bb};
        #pragma unroll
        for (int kt = 0; kt < 4; ++kt) {
            Frag16 a, b;
            int kb = kt*32 + 8*half;
            const U4* pa0 = (const U4*)&sO[m*128 + kb];
            const U4* pa1 = (const U4*)&sO[m*128 + kb + 16];
            a.q[0] = pa0[0]; a.q[1] = pa1[0];
            const U4* pb = (const U4*)&sW1[((kt*8 + nt) * 32 + lane) * 16];
            b.q[0] = pb[0]; b.q[1] = pb[1];
            c = wmma_bf16(a, b, c);
        }
        h[nt] = c;
    }
    __syncthreads();

    // relu(h) back to LDS (row-major bf16), C-layout scatter
    for (int nt = 0; nt < 8; ++nt) {
        #pragma unroll
        for (int v = 0; v < 8; ++v) {
            int mm = v + 8*half;
            sO[mm*128 + nt*16 + m] = f2bf(fmaxf(h[nt][v], 0.0f));
        }
    }
    __syncthreads();

    // GEMM2: y = h @ W2 + b2, then store / accumulate
    const bool fullRows = (rowBase + 16 <= N);
    for (int nt = 0; nt < 8; ++nt) {
        float bb = b2[nt*16 + m];
        v8f c = {bb,bb,bb,bb,bb,bb,bb,bb};
        #pragma unroll
        for (int kt = 0; kt < 4; ++kt) {
            Frag16 a, b;
            int kb = kt*32 + 8*half;
            const U4* pa0 = (const U4*)&sO[m*128 + kb];
            const U4* pa1 = (const U4*)&sO[m*128 + kb + 16];
            a.q[0] = pa0[0]; a.q[1] = pa1[0];
            const U4* pb = (const U4*)&sW2[((kt*8 + nt) * 32 + lane) * 16];
            b.q[0] = pb[0]; b.q[1] = pb[1];
            c = wmma_bf16(a, b, c);
        }
        if (fullRows) {
            #pragma unroll
            for (int v = 0; v < 8; ++v) {
                long  o   = (long)(rowBase + v + 8*half)*128 + nt*16 + m;
                float val = c[v];
                if (accumulate) val += out[o];
                if (doRelu)     val  = fmaxf(val, 0.0f);
                out[o] = val;
            }
        } else {
            #pragma unroll
            for (int v = 0; v < 8; ++v) {
                int r = rowBase + v + 8*half;
                if (r < N) {
                    long  o   = (long)r*128 + nt*16 + m;
                    float val = c[v];
                    if (accumulate) val += out[o];
                    if (doRelu)     val  = fmaxf(val, 0.0f);
                    out[o] = val;
                }
            }
        }
    }
}

// ---------------------------------------------------------------------------
extern "C" void kernel_launch(void* const* d_in, const int* in_sizes, int n_in,
                              void* d_out, int out_size, void* d_ws, size_t ws_size,
                              hipStream_t stream)
{
    (void)n_in; (void)out_size; (void)ws_size;

    const float* x_lig  = (const float*)d_in[0];
    const float* x_pock = (const float*)d_in[1];
    const int*   ei_b   = (const int*)  d_in[2];
    const float* ew_b   = (const float*)d_in[3];
    const float* ea_b   = (const float*)d_in[4];
    const int*   ei_l   = (const int*)  d_in[5];
    const float* ew_l   = (const float*)d_in[6];
    const float* ea_l   = (const float*)d_in[7];
    const int*   ei_p   = (const int*)  d_in[8];
    const float* ew_p   = (const float*)d_in[9];
    const float* ea_p   = (const float*)d_in[10];
    const float* W0     = (const float*)d_in[11];
    const float* Wd     = (const float*)d_in[12];
    const float* bd     = (const float*)d_in[13];
    const float* We     = (const float*)d_in[14];
    const float* be     = (const float*)d_in[15];
    const float* W1     = (const float*)d_in[16];
    const float* b1     = (const float*)d_in[17];
    const float* W2     = (const float*)d_in[18];
    const float* b2     = (const float*)d_in[19];

    const int N  = in_sizes[0] / 128;
    const int Eb = in_sizes[3];
    const int El = in_sizes[6];
    const int Ep = in_sizes[9];

    char* ws = (char*)d_ws;
    unsigned short* pWd = (unsigned short*)(ws);             // 6 * 32768 bf16
    unsigned short* pW1 = (unsigned short*)(ws + 393216);    // 6 * 16384 bf16
    unsigned short* pW2 = (unsigned short*)(ws + 589824);    // 6 * 16384 bf16
    float* agg = (float*)(ws + 786432);                      // N*128 f32
    float* xl1 = agg + (size_t)N * 128;                      // N*128 f32
    float* xp1 = xl1 + (size_t)N * 128;                      // N*128 f32

    pack_wmma_b<<<192, 256, 0, stream>>>(Wd, pWd, 128, 256, 6);
    pack_wmma_b<<< 96, 256, 0, stream>>>(W1, pW1, 128, 128, 6);
    pack_wmma_b<<< 96, 256, 0, stream>>>(W2, pW2, 128, 128, 6);

    const long nodeElems  = (long)N * 128;
    const int  nodeBlocks = (N + 127) / 128;
    const unsigned edgeLds = 65536 + 3072;   // Wd pack + edge metadata
    const unsigned nodeLds = 65536 + 32768;  // W1+W2 packs + o/h tiles

    auto conv = [&](int l, int t,
                    const float* ewp, const int* eip, const float* eap, int E,
                    const float* xsrc, const float* xdst, float* outp,
                    int accumulate, int doRelu)
    {
        zero_kernel<<<2048, 256, 0, stream>>>(agg, nodeElems);
        const int lt = l*3 + t;
        edge_conv_kernel<<<(E + 127) / 128, 256, edgeLds, stream>>>(
            ewp, eip, eap, xsrc,
            pWd + (size_t)lt * 32768, bd + lt*256,
            W0 + t*16, We + lt*512, be + lt*128, agg, E);
        node_mlp_kernel<<<nodeBlocks, 256, nodeLds, stream>>>(
            agg, xdst,
            pW1 + (size_t)lt * 16384, b1 + lt*128,
            pW2 + (size_t)lt * 16384, b2 + lt*128,
            outp, N, accumulate, doRelu);
    };

    float* outF = (float*)d_out;

    // layer 0 -> workspace (inter-layer ReLU)
    conv(0, 0, ew_b, ei_b, ea_b, Eb, x_lig,  x_lig,  xl1, /*acc=*/0, /*relu=*/0);
    conv(0, 2, ew_p, ei_p, ea_p, Ep, x_pock, x_lig,  xl1, /*acc=*/1, /*relu=*/1);
    conv(0, 1, ew_l, ei_l, ea_l, El, x_lig,  x_pock, xp1, /*acc=*/0, /*relu=*/1);

    // layer 1 -> d_out (no final ReLU); lig rows then pock rows
    conv(1, 0, ew_b, ei_b, ea_b, Eb, xl1, xl1, outF,                  0, 0);
    conv(1, 2, ew_p, ei_p, ea_p, Ep, xp1, xl1, outF,                  1, 0);
    conv(1, 1, ew_l, ei_l, ea_l, El, xl1, xp1, outF + (size_t)N*128,  0, 0);
}
